// MILModel_79078937854181
// MI455X (gfx1250) — compile-verified, matrix operand order checked by reading, not run on hardware
//
#include <hip/hip_runtime.h>

// ---------------------------------------------------------------------------
// MIL Nystromformer forward for MI455X (gfx1250).
// GEMMs run on v_wmma_f32_16x16x32_f16 (wave32), A tiles staged to LDS with
// global_load_async_to_lds_b128 + s_wait_asynccnt (CDNA5 async path).
// ---------------------------------------------------------------------------

typedef __attribute__((ext_vector_type(16))) _Float16 v16h;
typedef __attribute__((ext_vector_type(8)))  float    v8f;

constexpr int CB     = 2;       // batch
constexpr int CN     = 8192;    // sequence
constexpr int CBN    = CB * CN;
constexpr int CFD    = 1024;    // input feature dim
constexpr int CH     = 512;     // model dim
constexpr int CH3    = 1536;
constexpr int CHEADS = 8;
constexpr int CDH    = 64;      // head dim
constexpr int CM     = 256;     // landmarks
constexpr int CNI    = 5;       // decoder queries
constexpr int CNPAD  = 256;     // padded decoder seq
constexpr int CBH    = CB * CHEADS;  // 16
constexpr float CSCALE = 0.125f;     // DH^-0.5

// ===========================================================================
// Batched WMMA GEMM, block tile 32(M) x 256(N), 8 waves (2x4), 4 N-tiles/wave.
//   C[M,N] = act(scale * A[M,K] @ op(B) + bias)
//   A fp32 row-major (lda). transB=1: B is [N,K] row-major -> A@B^T
//                           transB=0: B is [K,N] row-major -> A@B
// Requirements (all call sites satisfy): M%32==0, K%32==0. N guarded per tile.
// A 32x32 fp32 tile is async-staged into LDS (double buffered) and shared by
// all 8 waves; each wave reuses its A fragment across 4 WMMAs.
// ===========================================================================
__global__ void __launch_bounds__(256)
gemm_wmma_kernel(const float* __restrict__ A, const float* __restrict__ Bm,
                 const float* __restrict__ bias, float* __restrict__ C,
                 int Mdim, int Ndim, int Kdim,
                 long long sA, long long sB, long long sC,
                 int lda, int ldb, int ldc,
                 float outScale, int transB, int relu)
{
    __shared__ float aTile[2][32][33];   // +1 pad to spread LDS banks

    const int batch = blockIdx.z;
    const float* Ab = A  + (long long)batch * sA;
    const float* Bb = Bm + (long long)batch * sB;
    float*       Cb = C  + (long long)batch * sC;
    (void)Mdim;

    const int wave  = threadIdx.x >> 5;
    const int lane  = threadIdx.x & 31;
    const int waveM = wave >> 2;                       // 0..1
    const int waveN = wave & 3;                        // 0..3
    const int mBase = blockIdx.y * 32;                 // exact (M % 32 == 0)
    const int nBase = blockIdx.x * 256 + waveN * 64;   // 4 N-tiles per wave

    const int hi = lane >> 4;
    const int lo = lane & 15;

    // per-thread async staging coords: one b128 each covers the 32x32 tile
    const int sRow = threadIdx.x >> 3;        // 0..31
    const int sCol = (threadIdx.x & 7) * 4;   // 0,4,...,28 (floats)
    const float* aStageBase = Ab + (long long)(mBase + sRow) * lda + sCol;

    {   // prologue: stage k0 = 0 into buffer 0
        unsigned dst = (unsigned)(unsigned long long)(const void*)&aTile[0][sRow][sCol];
        asm volatile("global_load_async_to_lds_b128 %0, %1, off"
                     :: "v"(dst), "v"(aStageBase) : "memory");
    }

    v8f acc0 = {}, acc1 = {}, acc2 = {}, acc3 = {};
    int cbuf = 0;
    for (int k0 = 0; k0 < Kdim; k0 += 32) {
        asm volatile("s_wait_asynccnt 0" ::: "memory");
        __syncthreads();
        if (k0 + 32 < Kdim) {   // stage next K-slab into the other buffer
            unsigned dst = (unsigned)(unsigned long long)(const void*)&aTile[cbuf ^ 1][sRow][sCol];
            const float* src = aStageBase + k0 + 32;
            asm volatile("global_load_async_to_lds_b128 %0, %1, off"
                         :: "v"(dst), "v"(src) : "memory");
        }
        // A fragment from LDS (ISA 16-bit A layout: K = 8*hi+{0..7}, +16)
        v16h af;
        {
            const float* ar = &aTile[cbuf][waveM * 16 + lo][8 * hi];
#pragma unroll
            for (int e = 0; e < 8; ++e) af[e]     = (_Float16)ar[e];
#pragma unroll
            for (int e = 0; e < 8; ++e) af[e + 8] = (_Float16)ar[e + 16];
        }
#pragma unroll
        for (int tn = 0; tn < 4; ++tn) {
            const int n0 = nBase + tn * 16;
            if (n0 >= Ndim) continue;           // wave-uniform guard
            v16h bf;
            if (transB) {
                const float* bp = Bb + (long long)(n0 + lo) * ldb + k0 + 8 * hi;
                __builtin_prefetch(bp + 32, 0, 1);
#pragma unroll
                for (int e = 0; e < 8; ++e) bf[e]     = (_Float16)bp[e];
#pragma unroll
                for (int e = 0; e < 8; ++e) bf[e + 8] = (_Float16)bp[e + 16];
            } else {
                const float* bp = Bb + (long long)(k0 + 8 * hi) * ldb + n0 + lo;
#pragma unroll
                for (int e = 0; e < 8; ++e) bf[e]     = (_Float16)bp[(long long)e * ldb];
                const float* bp2 = bp + (long long)16 * ldb;
#pragma unroll
                for (int e = 0; e < 8; ++e) bf[e + 8] = (_Float16)bp2[(long long)e * ldb];
            }
            v8f& acc = tn == 0 ? acc0 : tn == 1 ? acc1 : tn == 2 ? acc2 : acc3;
            acc = __builtin_amdgcn_wmma_f32_16x16x32_f16(false, af, false, bf,
                                                         (short)0, acc, false, false);
        }
        cbuf ^= 1;
    }

#pragma unroll
    for (int tn = 0; tn < 4; ++tn) {
        const int n0 = nBase + tn * 16;
        if (n0 >= Ndim) continue;
        const v8f& acc = tn == 0 ? acc0 : tn == 1 ? acc1 : tn == 2 ? acc2 : acc3;
        const int nr = n0 + lo;
        const float bv = bias ? bias[nr] : 0.0f;
#pragma unroll
        for (int v = 0; v < 8; ++v) {
            const int cm = mBase + waveM * 16 + v + 8 * hi;   // C layout per ISA
            float r = acc[v] * outScale + bv;
            if (relu) r = fmaxf(r, 0.0f);
            Cb[(long long)cm * ldc + nr] = r;
        }
    }
}

// ===========================================================================
// Elementwise / reduction helper kernels
// ===========================================================================

// LayerNorm over H=512, optional residual b. Row via blockIdx.x, batch via .y.
__global__ void __launch_bounds__(256)
ln_kernel(const float* __restrict__ a, long long aBatch,
          const float* __restrict__ b, long long bBatch,
          const float* __restrict__ g, const float* __restrict__ be,
          float* __restrict__ out, long long oBatch)
{
    const int row = blockIdx.x, batch = blockIdx.y, t = threadIdx.x;
    const float* ap = a + (long long)batch * aBatch + (long long)row * CH;
    const float* bp = b ? (b + (long long)batch * bBatch + (long long)row * CH) : nullptr;
    float* op = out + (long long)batch * oBatch + (long long)row * CH;
    __shared__ float red[256];
    float v0 = ap[t]       + (bp ? bp[t]       : 0.f);
    float v1 = ap[t + 256] + (bp ? bp[t + 256] : 0.f);
    red[t] = v0 + v1; __syncthreads();
    for (int o = 128; o > 0; o >>= 1) { if (t < o) red[t] += red[t + o]; __syncthreads(); }
    const float mu = red[0] * (1.0f / CH); __syncthreads();
    const float d0 = v0 - mu, d1 = v1 - mu;
    red[t] = d0 * d0 + d1 * d1; __syncthreads();
    for (int o = 128; o > 0; o >>= 1) { if (t < o) red[t] += red[t + o]; __syncthreads(); }
    const float rs = rsqrtf(red[0] * (1.0f / CH) + 1e-5f);
    op[t]       = d0 * rs * g[t]       + be[t];
    op[t + 256] = d1 * rs * g[t + 256] + be[t + 256];
}

// RoPE as in reference: x*cos(a) + roll(x,1,seq)*sin(a), a = i * 10000^{-(d%256)/256}
__global__ void rope_kernel(const float* __restrict__ in, float* __restrict__ out, int n)
{
    long long idx = (long long)blockIdx.x * 256 + threadIdx.x;
    if (idx >= (long long)CB * n * CH) return;
    const int d = (int)(idx % CH);
    long long r = idx / CH;
    const int i = (int)(r % n);
    const int b = (int)(r / n);
    const int j = d & 255;
    const float ang = (float)i * powf(10000.0f, -(float)j / 256.0f);
    const int ip = (i == 0) ? (n - 1) : (i - 1);
    const float xv = in[idx];
    const float xr = in[((long long)b * n + ip) * CH + d];
    out[idx] = xv * cosf(ang) + xr * sinf(ang);
}

// qkv [b,n,3H] -> q,k,v heads [b,h,n,dh]; q scaled.
__global__ void split_qkv_kernel(const float* __restrict__ qkv,
                                 float* __restrict__ q, float* __restrict__ k,
                                 float* __restrict__ v, int n, float qscale)
{
    long long idx = (long long)blockIdx.x * 256 + threadIdx.x;
    if (idx >= (long long)CB * n * CH) return;
    const int c = (int)(idx % CH);
    long long r = idx / CH;
    const int i = (int)(r % n);
    const int b = (int)(r / n);
    const int h = c / CDH, d = c % CDH;
    const long long src = ((long long)b * n + i) * CH3;
    const long long dst = (((long long)b * CHEADS + h) * n + i) * CDH + d;
    q[dst] = qkv[src + c] * qscale;
    k[dst] = qkv[src + CH + c];
    v[dst] = qkv[src + 2 * CH + c];
}

// landmark mean over l contiguous rows: [bh,n,dh] -> [bh,M,dh]
__global__ void landmark_mean_kernel(const float* __restrict__ q,
                                     float* __restrict__ ql, int n, int l)
{
    int idx = blockIdx.x * 256 + threadIdx.x;
    if (idx >= CBH * CM * CDH) return;
    const int d  = idx % CDH;
    const int r  = idx / CDH;
    const int m  = r % CM;
    const int bh = r / CM;
    const float* base = q + ((long long)bh * n + (long long)m * l) * CDH + d;
    float s = 0.f;
    for (int j = 0; j < l; ++j) s += base[(long long)j * CDH];
    ql[idx] = s / (float)l;
}

// In-place stable row softmax, one block per row of length L.
__global__ void __launch_bounds__(256)
softmax_rows_kernel(float* __restrict__ x, int L)
{
    float* p = x + (long long)blockIdx.x * L;
    __shared__ float red[256];
    const int t = threadIdx.x;
    float mx = -3.4e38f;
    for (int j = t; j < L; j += 256) mx = fmaxf(mx, p[j]);
    red[t] = mx; __syncthreads();
    for (int o = 128; o > 0; o >>= 1) { if (t < o) red[t] = fmaxf(red[t], red[t + o]); __syncthreads(); }
    mx = red[0]; __syncthreads();
    float s = 0.f;
    for (int j = t; j < L; j += 256) { float e = expf(p[j] - mx); p[j] = e; s += e; }
    red[t] = s; __syncthreads();
    for (int o = 128; o > 0; o >>= 1) { if (t < o) red[t] += red[t + o]; __syncthreads(); }
    const float inv = 1.0f / red[0];
    for (int j = t; j < L; j += 256) p[j] *= inv;
}

__global__ void zero_u32_kernel(unsigned* p, int n)
{
    int i = blockIdx.x * 256 + threadIdx.x;
    if (i < n) p[i] = 0u;
}

// max over (bh,rows) of sum_j |x|: trans=0 -> abs.sum(-1).max -> scal[0]
//                                 trans=1 -> abs.sum(-2).max -> scal[1]
__global__ void pinv_norm_kernel(const float* __restrict__ x, unsigned* scal, int trans)
{
    const int bh = blockIdx.y;
    const int t = threadIdx.x;
    const float* m = x + (long long)bh * CM * CM;
    if (!trans) {
        const int row = blockIdx.x;
        __shared__ float red[256];
        red[t] = fabsf(m[row * CM + t]); __syncthreads();
        for (int o = 128; o > 0; o >>= 1) { if (t < o) red[t] += red[t + o]; __syncthreads(); }
        if (t == 0) atomicMax(scal + 0, __float_as_uint(red[0]));
    } else {
        float s = 0.f;
        for (int r = 0; r < CM; ++r) s += fabsf(m[r * CM + t]);
        atomicMax(scal + 1, __float_as_uint(s));
    }
}

// z = x^T / (col * row)
__global__ void pinv_init_z_kernel(const float* __restrict__ x,
                                   const unsigned* __restrict__ scal,
                                   float* __restrict__ z)
{
    int idx = blockIdx.x * 256 + threadIdx.x;
    if (idx >= CBH * CM * CM) return;
    const int j  = idx % CM;
    const int i  = (idx / CM) % CM;
    const int bh = idx / (CM * CM);
    const float inv = 1.0f / (__uint_as_float(scal[0]) * __uint_as_float(scal[1]));
    z[idx] = x[(long long)bh * CM * CM + j * CM + i] * inv;
}

// t = alpha*I - s  (in-place safe)
__global__ void alpha_i_minus_kernel(const float* __restrict__ s,
                                     float* __restrict__ t, float alpha)
{
    int idx = blockIdx.x * 256 + threadIdx.x;
    if (idx >= CBH * CM * CM) return;
    const int j = idx % CM;
    const int i = (idx / CM) % CM;
    t[idx] = ((i == j) ? alpha : 0.0f) - s[idx];
}

// depthwise residual conv over seq (K=33, zero pad 16), accumulated into out.
__global__ void conv_res_kernel(const float* __restrict__ v,
                                const float* __restrict__ w,
                                float* __restrict__ out, int n)
{
    long long idx = (long long)blockIdx.x * 256 + threadIdx.x;
    if (idx >= (long long)CBH * n * CDH) return;
    const int d = (int)(idx % CDH);
    long long r = idx / CDH;
    const int i  = (int)(r % n);
    const int bh = (int)(r / n);
    const int h = bh % CHEADS;
    const float* wp = w + h * 33;
    const float* vb = v + ((long long)bh * n) * CDH + d;
    float s = 0.f;
    for (int kk = 0; kk < 33; ++kk) {
        const int ii = i + kk - 16;
        if (ii >= 0 && ii < n) s += wp[kk] * vb[(long long)ii * CDH];
    }
    out[idx] += s;
}

// [b,h,n,dh] -> [b,n,h*dh]
__global__ void merge_heads_kernel(const float* __restrict__ oh,
                                   float* __restrict__ out, int n)
{
    long long idx = (long long)blockIdx.x * 256 + threadIdx.x;
    if (idx >= (long long)CB * n * CH) return;
    const int c = (int)(idx % CH);
    long long r = idx / CH;
    const int i = (int)(r % n);
    const int b = (int)(r / n);
    const int h = c / CDH, d = c % CDH;
    out[idx] = oh[(((long long)b * CHEADS + h) * n + i) * CDH + d];
}

// pad hs[b,5,H] at the FRONT to [b,256,H]
__global__ void pad_front_kernel(const float* __restrict__ hs, float* __restrict__ xp)
{
    int idx = blockIdx.x * 256 + threadIdx.x;
    if (idx >= CB * CNPAD * CH) return;
    const int c = idx % CH;
    const int r = idx / CH;
    const int i = r % CNPAD;
    const int b = r / CNPAD;
    xp[idx] = (i >= CNPAD - CNI) ? hs[((long long)b * CNI + (i - (CNPAD - CNI))) * CH + c] : 0.0f;
}

__global__ void bcast_query_kernel(const float* __restrict__ q, float* __restrict__ hs)
{
    int idx = blockIdx.x * 256 + threadIdx.x;
    if (idx >= CB * CNI * CH) return;
    const int c = idx % CH;
    const int i = (idx / CH) % CNI;
    hs[idx] = q[i * CH + c];
}

// out[r,o] = act(in[r,:] . w[o,:] + bias[o])  (small irregular GEMMs)
__global__ void small_linear_kernel(const float* __restrict__ in,
                                    const float* __restrict__ w,
                                    const float* __restrict__ bias,
                                    float* __restrict__ out,
                                    int rows, int nout, int kdim, int relu)
{
    int idx = blockIdx.x * 256 + threadIdx.x;
    if (idx >= rows * nout) return;
    const int o = idx % nout;
    const int r = idx / nout;
    const float* ip = in + (long long)r * kdim;
    const float* wp = w + (long long)o * kdim;
    float s = bias ? bias[o] : 0.0f;
    for (int k = 0; k < kdim; ++k) s += ip[k] * wp[k];
    out[idx] = relu ? fmaxf(s, 0.0f) : s;
}

// cross-attn scores + softmax for 5 query rows: block per (b,h,qi).
__global__ void __launch_bounds__(256)
cross_score_kernel(const float* __restrict__ q, const float* __restrict__ k,
                   float* __restrict__ sbuf)
{
    const int blk = blockIdx.x;                 // (b*HEADS + h)*NI + qi
    const int qi = blk % CNI;
    const int h  = (blk / CNI) % CHEADS;
    const int b  = blk / (CNI * CHEADS);
    __shared__ float qd[CDH];
    __shared__ float red[256];
    const int t = threadIdx.x;
    if (t < CDH) qd[t] = q[((long long)b * CNI + qi) * CH + h * CDH + t];
    __syncthreads();
    float* sp = sbuf + (long long)blk * CN;
    const float* kb = k + (long long)b * CN * CH + h * CDH;
    float mx = -3.4e38f;
    for (int j = t; j < CN; j += 256) {
        const float* kp = kb + (long long)j * CH;
        float s = 0.f;
#pragma unroll
        for (int d = 0; d < CDH; ++d) s += qd[d] * kp[d];
        s *= CSCALE;
        sp[j] = s;
        mx = fmaxf(mx, s);
    }
    red[t] = mx; __syncthreads();
    for (int o = 128; o > 0; o >>= 1) { if (t < o) red[t] = fmaxf(red[t], red[t + o]); __syncthreads(); }
    mx = red[0]; __syncthreads();
    float ssum = 0.f;
    for (int j = t; j < CN; j += 256) { float e = expf(sp[j] - mx); sp[j] = e; ssum += e; }
    red[t] = ssum; __syncthreads();
    for (int o = 128; o > 0; o >>= 1) { if (t < o) red[t] += red[t + o]; __syncthreads(); }
    const float inv = 1.0f / red[0];
    for (int j = t; j < CN; j += 256) sp[j] *= inv;
}

// out[b,qi,h*64+d] = sum_j p[b,h,qi,j] * v[b,j,h*64+d]
__global__ void cross_attv_kernel(const float* __restrict__ p,
                                  const float* __restrict__ v,
                                  float* __restrict__ out)
{
    int idx = blockIdx.x * 256 + threadIdx.x;
    if (idx >= CB * CNI * CH) return;
    const int c  = idx % CH;
    const int r  = idx / CH;
    const int qi = r % CNI;
    const int b  = r / CNI;
    const int h = c / CDH;
    const float* pp = p + (((long long)b * CHEADS + h) * CNI + qi) * CN;
    const float* vp = v + (long long)b * CN * CH + c;
    float s = 0.f;
    for (int j = 0; j < CN; ++j) s += pp[j] * vp[(long long)j * CH];
    out[idx] = s;
}

// attention pooling over NI + classifier. One block (512 thr) per batch.
__global__ void __launch_bounds__(512)
pool_cls_kernel(const float* __restrict__ hs, const float* __restrict__ aw_w,
                const float* __restrict__ aw_b, const float* __restrict__ cls_w,
                const float* __restrict__ cls_b, float* __restrict__ out)
{
    const int b = blockIdx.x;
    const int t = threadIdx.x;
    __shared__ float wv[CNI];
    __shared__ float pooled[CH];
    if (t < CNI) {
        const float* hp = hs + ((long long)b * CNI + t) * CH;
        float s = aw_b[0];
        for (int k = 0; k < CH; ++k) s += hp[k] * aw_w[k];
        wv[t] = s;
    }
    __syncthreads();
    if (t == 0) {
        float mx = wv[0];
        for (int i = 1; i < CNI; ++i) mx = fmaxf(mx, wv[i]);
        float ss = 0.f;
        for (int i = 0; i < CNI; ++i) { wv[i] = expf(wv[i] - mx); ss += wv[i]; }
        for (int i = 0; i < CNI; ++i) wv[i] /= ss;
    }
    __syncthreads();
    float s = 0.f;
    for (int i = 0; i < CNI; ++i) s += hs[((long long)b * CNI + i) * CH + t] * wv[i];
    pooled[t] = s;
    __syncthreads();
    if (t < CNI) {
        float r = cls_b[t];
        const float* cw = cls_w + t * CH;
        for (int k = 0; k < CH; ++k) r += pooled[k] * cw[k];
        out[b * CNI + t] = r;
    }
}

// ===========================================================================
// Host orchestration
// ===========================================================================
extern "C" void kernel_launch(void* const* d_in, const int* in_sizes, int n_in,
                              void* d_out, int out_size, void* d_ws, size_t ws_size,
                              hipStream_t stream)
{
    (void)in_sizes; (void)n_in; (void)out_size; (void)ws_size;

    // Input map (setup_inputs insertion-order, depth-first):
    // 0 x | 1 fc_w 2 fc_b 3 ln0_g 4 ln0_b
    // enc[0]: 5 qkv_w 6 out_w 7 out_b 8 res_w 9 w1 10 b1 11 w2 12 b2
    //         13 n1_g 14 n1_b 15 n2_g 16 n2_b
    // dec[L] (base 17, 35): +0 qkv_w +1 out_w +2 out_b +3 res_w +4 w1 +5 b1
    //         +6 w2 +7 b2 +8 n1_g +9 n1_b +10 n2_g +11 n2_b
    //         +12 ca.in_w +13 ca.in_b +14 ca.out_w +15 ca.out_b +16 n3_g +17 n3_b
    // 53 query 54 aw_w 55 aw_b 56 cls_w 57 cls_b
    auto P = [&](int i) { return (const float*)d_in[i]; };

    float* W = (float*)d_ws;
    size_t cur = 0;
    auto take = [&](size_t n) { size_t r = cur; cur += n; return r; };

    const long long MM = (long long)CM * CM;
    const size_t oX    = take((size_t)CBN * CH);
    const size_t oT1   = take((size_t)CBN * CH);
    const size_t oT2   = take((size_t)CBN * CH);
    const size_t oMEM  = take((size_t)CBN * CH);
    const size_t oQKV  = take((size_t)CBN * CH3);   // reused for cross K/V
    const size_t oQ    = take((size_t)CBN * CH);
    const size_t oK    = take((size_t)CBN * CH);
    const size_t oV    = take((size_t)CBN * CH);
    const size_t oA1   = take((size_t)CBH * CN * CM);
    const size_t oA3   = take((size_t)CBH * CM * CN);
    const size_t oQL   = take((size_t)CBH * CM * CDH);
    const size_t oKL   = take((size_t)CBH * CM * CDH);
    const size_t oA2   = take((size_t)CBH * MM);
    const size_t oZ    = take((size_t)CBH * MM);
    const size_t oZB   = take((size_t)CBH * MM);
    const size_t oXZ   = take((size_t)CBH * MM);
    const size_t oTTa  = take((size_t)CBH * MM);
    const size_t oTTb  = take((size_t)CBH * MM);
    const size_t oA3V  = take((size_t)CBH * CM * CDH);
    const size_t oSC   = take(16);
    const size_t oHS   = take((size_t)CB * CNI * CH);
    const size_t oHPAD = take((size_t)CB * CNPAD * CH);
    const size_t oDQKV = take((size_t)CB * CNPAD * CH3);
    const size_t oDQ   = take((size_t)CBH * CNPAD * CDH);
    const size_t oDK   = take((size_t)CBH * CNPAD * CDH);
    const size_t oDV   = take((size_t)CBH * CNPAD * CDH);
    const size_t oDA1  = take((size_t)CBH * CNPAD * CM);
    const size_t oDA3  = take((size_t)CBH * CM * CNPAD);
    const size_t oDOH  = take((size_t)CBH * CNPAD * CDH);
    const size_t oDM   = take((size_t)CB * CNPAD * CH);
    const size_t oDP   = take((size_t)CB * CNPAD * CH);
    const size_t oCQ   = take((size_t)CB * CNI * CH);
    const size_t oCS   = take((size_t)CBH * CNI * CN);
    const size_t oCO   = take((size_t)CB * CNI * CH);
    const size_t oCP   = take((size_t)CB * CNI * CH);
    const size_t oDH1  = take((size_t)CB * CNI * CH);
    const size_t oDH2  = take((size_t)CB * CNI * CH);
    const size_t oDF1  = take((size_t)CB * CNI * CH);
    const size_t oDF2  = take((size_t)CB * CNI * CH);
    const size_t oCK = oQKV;                         // alias into freed QKV
    const size_t oCV = oQKV + (size_t)CBN * CH;

    auto blocks = [](long long n) { return (unsigned)((n + 255) / 256); };

    auto gemm = [&](const float* Ap, const float* Bp, const float* biasp, float* Cp,
                    int Md, int Nd, int Kd, long long sA, long long sB, long long sC,
                    int lda, int ldb, int ldc, float sc, int tb, int rl, int batch) {
        dim3 g((unsigned)((Nd + 255) / 256), (unsigned)(Md / 32), (unsigned)batch);
        gemm_wmma_kernel<<<g, 256, 0, stream>>>(Ap, Bp, biasp, Cp, Md, Nd, Kd,
                                                sA, sB, sC, lda, ldb, ldc, sc, tb, rl);
    };

    // Moore-Penrose iterative pinv of the 16 batched [256,256] matrices in xmat.
    auto run_pinv = [&](const float* xmat) -> const float* {
        zero_u32_kernel<<<1, 256, 0, stream>>>((unsigned*)(W + oSC), 16);
        pinv_norm_kernel<<<dim3(CM, CBH), 256, 0, stream>>>(xmat, (unsigned*)(W + oSC), 0);
        pinv_norm_kernel<<<dim3(1, CBH), 256, 0, stream>>>(xmat, (unsigned*)(W + oSC), 1);
        pinv_init_z_kernel<<<blocks((long long)CBH * MM), 256, 0, stream>>>(
            xmat, (const unsigned*)(W + oSC), W + oZ);
        float* zc = W + oZ; float* zn = W + oZB;
        for (int it = 0; it < 6; ++it) {
            gemm(xmat, zc, nullptr, W + oXZ, CM, CM, CM, MM, MM, MM, CM, CM, CM, 1.f, 0, 0, CBH);
            alpha_i_minus_kernel<<<blocks((long long)CBH * MM), 256, 0, stream>>>(W + oXZ, W + oTTa, 7.f);
            gemm(W + oXZ, W + oTTa, nullptr, W + oTTb, CM, CM, CM, MM, MM, MM, CM, CM, CM, 1.f, 0, 0, CBH);
            alpha_i_minus_kernel<<<blocks((long long)CBH * MM), 256, 0, stream>>>(W + oTTb, W + oTTb, 15.f);
            gemm(W + oXZ, W + oTTb, nullptr, W + oTTa, CM, CM, CM, MM, MM, MM, CM, CM, CM, 1.f, 0, 0, CBH);
            alpha_i_minus_kernel<<<blocks((long long)CBH * MM), 256, 0, stream>>>(W + oTTa, W + oTTa, 13.f);
            gemm(zc, W + oTTa, nullptr, zn, CM, CM, CM, MM, MM, MM, CM, CM, CM, 0.25f, 0, 0, CBH);
            float* ts = zc; zc = zn; zn = ts;
        }
        return zc;   // 6 swaps -> back at W+oZ
    };

    // ------------------------- stem -------------------------
    gemm(P(0), P(1), P(2), W + oX, CBN, CH, CFD, 0, 0, 0, CFD, CFD, CH, 1.f, 1, 0, 1);
    ln_kernel<<<dim3(CBN, 1), 256, 0, stream>>>(W + oX, 0, nullptr, 0, P(3), P(4), W + oT1, 0);
    rope_kernel<<<blocks((long long)CBN * CH), 256, 0, stream>>>(W + oT1, W + oMEM, CN);

    // ------------------------- encoder layer (Nystrom) -------------------------
    {
        const float *qkv_w = P(5), *out_w = P(6), *out_b = P(7), *res_w = P(8);
        const float *w1 = P(9), *b1 = P(10), *w2 = P(11), *b2 = P(12);
        const float *n1g = P(13), *n1b = P(14), *n2g = P(15), *n2b = P(16);

        gemm(W + oMEM, qkv_w, nullptr, W + oQKV, CBN, CH3, CH, 0, 0, 0, CH, CH, CH3, 1.f, 1, 0, 1);
        split_qkv_kernel<<<blocks((long long)CBN * CH), 256, 0, stream>>>(
            W + oQKV, W + oQ, W + oK, W + oV, CN, CSCALE);
        landmark_mean_kernel<<<blocks((long long)CBH * CM * CDH), 256, 0, stream>>>(W + oQ, W + oQL, CN, CN / CM);
        landmark_mean_kernel<<<blocks((long long)CBH * CM * CDH), 256, 0, stream>>>(W + oK, W + oKL, CN, CN / CM);

        gemm(W + oQ, W + oKL, nullptr, W + oA1, CN, CM, CDH,
             (long long)CN * CDH, (long long)CM * CDH, (long long)CN * CM, CDH, CDH, CM, 1.f, 1, 0, CBH);
        softmax_rows_kernel<<<CBH * CN, 256, 0, stream>>>(W + oA1, CM);
        gemm(W + oQL, W + oKL, nullptr, W + oA2, CM, CM, CDH,
             (long long)CM * CDH, (long long)CM * CDH, MM, CDH, CDH, CM, 1.f, 1, 0, CBH);
        softmax_rows_kernel<<<CBH * CM, 256, 0, stream>>>(W + oA2, CM);
        gemm(W + oQL, W + oK, nullptr, W + oA3, CM, CN, CDH,
             (long long)CM * CDH, (long long)CN * CDH, (long long)CM * CN, CDH, CDH, CN, 1.f, 1, 0, CBH);
        softmax_rows_kernel<<<CBH * CM, 256, 0, stream>>>(W + oA3, CN);

        const float* zf = run_pinv(W + oA2);

        gemm(W + oA3, W + oV, nullptr, W + oA3V, CM, CDH, CN,
             (long long)CM * CN, (long long)CN * CDH, (long long)CM * CDH, CN, CDH, CDH, 1.f, 0, 0, CBH);
        gemm(W + oA1, zf, nullptr, W + oA3, CN, CM, CM,                 // a1@pinv -> A3 region
             (long long)CN * CM, MM, (long long)CN * CM, CM, CM, CM, 1.f, 0, 0, CBH);
        gemm(W + oA3, W + oA3V, nullptr, W + oX, CN, CDH, CM,           // (a1@pinv)@(a3@v)
             (long long)CN * CM, (long long)CM * CDH, (long long)CN * CDH, CM, CDH, CDH, 1.f, 0, 0, CBH);
        conv_res_kernel<<<blocks((long long)CBH * CN * CDH), 256, 0, stream>>>(W + oV, res_w, W + oX, CN);
        merge_heads_kernel<<<blocks((long long)CBN * CH), 256, 0, stream>>>(W + oX, W + oT1, CN);
        gemm(W + oT1, out_w, out_b, W + oT2, CBN, CH, CH, 0, 0, 0, CH, CH, CH, 1.f, 1, 0, 1);

        ln_kernel<<<dim3(CBN, 1), 256, 0, stream>>>(W + oMEM, 0, W + oT2, 0, n1g, n1b, W + oX, 0);
        gemm(W + oX, w1, b1, W + oT1, CBN, CH, CH, 0, 0, 0, CH, CH, CH, 1.f, 1, 1, 1);
        gemm(W + oT1, w2, b2, W + oT2, CBN, CH, CH, 0, 0, 0, CH, CH, CH, 1.f, 1, 0, 1);
        ln_kernel<<<dim3(CBN, 1), 256, 0, stream>>>(W + oX, 0, W + oT2, 0, n2g, n2b, W + oMEM, 0);
    }

    // ------------------------- decoder -------------------------
    bcast_query_kernel<<<blocks((long long)CB * CNI * CH), 256, 0, stream>>>(P(53), W + oHS);

    const int decBase[2] = {17, 35};
    for (int L = 0; L < 2; ++L) {
        const int bx = decBase[L];
        const float *qkv_w = P(bx + 0), *out_w = P(bx + 1), *out_b = P(bx + 2), *res_w = P(bx + 3);
        const float *w1 = P(bx + 4), *b1 = P(bx + 5), *w2 = P(bx + 6), *b2 = P(bx + 7);
        const float *n1g = P(bx + 8), *n1b = P(bx + 9), *n2g = P(bx + 10), *n2b = P(bx + 11);
        const float *in_w = P(bx + 12), *in_b = P(bx + 13), *cow = P(bx + 14), *cob = P(bx + 15);
        const float *n3g = P(bx + 16), *n3b = P(bx + 17);

        // Nystrom self-attn on 5 queries padded to 256 at front
        pad_front_kernel<<<blocks((long long)CB * CNPAD * CH), 256, 0, stream>>>(W + oHS, W + oHPAD);
        gemm(W + oHPAD, qkv_w, nullptr, W + oDQKV, CB * CNPAD, CH3, CH, 0, 0, 0, CH, CH, CH3, 1.f, 1, 0, 1);
        split_qkv_kernel<<<blocks((long long)CB * CNPAD * CH), 256, 0, stream>>>(
            W + oDQKV, W + oDQ, W + oDK, W + oDV, CNPAD, CSCALE);
        landmark_mean_kernel<<<blocks((long long)CBH * CM * CDH), 256, 0, stream>>>(W + oDQ, W + oQL, CNPAD, 1);
        landmark_mean_kernel<<<blocks((long long)CBH * CM * CDH), 256, 0, stream>>>(W + oDK, W + oKL, CNPAD, 1);

        gemm(W + oDQ, W + oKL, nullptr, W + oDA1, CNPAD, CM, CDH,
             (long long)CNPAD * CDH, (long long)CM * CDH, (long long)CNPAD * CM, CDH, CDH, CM, 1.f, 1, 0, CBH);
        softmax_rows_kernel<<<CBH * CNPAD, 256, 0, stream>>>(W + oDA1, CM);
        gemm(W + oQL, W + oKL, nullptr, W + oA2, CM, CM, CDH,
             (long long)CM * CDH, (long long)CM * CDH, MM, CDH, CDH, CM, 1.f, 1, 0, CBH);
        softmax_rows_kernel<<<CBH * CM, 256, 0, stream>>>(W + oA2, CM);
        gemm(W + oQL, W + oDK, nullptr, W + oDA3, CM, CNPAD, CDH,
             (long long)CM * CDH, (long long)CNPAD * CDH, (long long)CM * CNPAD, CDH, CDH, CNPAD, 1.f, 1, 0, CBH);
        softmax_rows_kernel<<<CBH * CM, 256, 0, stream>>>(W + oDA3, CNPAD);

        const float* zf = run_pinv(W + oA2);

        gemm(W + oDA3, W + oDV, nullptr, W + oA3V, CM, CDH, CNPAD,
             (long long)CM * CNPAD, (long long)CNPAD * CDH, (long long)CM * CDH, CNPAD, CDH, CDH, 1.f, 0, 0, CBH);
        gemm(W + oDA1, zf, nullptr, W + oXZ, CNPAD, CM, CM,
             (long long)CNPAD * CM, MM, (long long)CNPAD * CM, CM, CM, CM, 1.f, 0, 0, CBH);
        gemm(W + oXZ, W + oA3V, nullptr, W + oDOH, CNPAD, CDH, CM,
             (long long)CNPAD * CM, (long long)CM * CDH, (long long)CNPAD * CDH, CM, CDH, CDH, 1.f, 0, 0, CBH);
        conv_res_kernel<<<blocks((long long)CBH * CNPAD * CDH), 256, 0, stream>>>(W + oDV, res_w, W + oDOH, CNPAD);
        merge_heads_kernel<<<blocks((long long)CB * CNPAD * CH), 256, 0, stream>>>(W + oDOH, W + oDM, CNPAD);
        gemm(W + oDM, out_w, out_b, W + oDP, CB * CNPAD, CH, CH, 0, 0, 0, CH, CH, CH, 1.f, 1, 0, 1);

        // h = LN(hs + attn_out[:, -5:])
        ln_kernel<<<dim3(CNI, CB), 256, 0, stream>>>(
            W + oDP + (long long)(CNPAD - CNI) * CH, (long long)CNPAD * CH,
            W + oHS, (long long)CNI * CH, n1g, n1b, W + oDH1, (long long)CNI * CH);

        // cross attention to encoder memory
        small_linear_kernel<<<blocks((long long)CB * CNI * CH), 256, 0, stream>>>(
            W + oDH1, in_w, in_b, W + oCQ, CB * CNI, CH, CH, 0);
        gemm(W + oMEM, in_w + (long long)CH * CH, in_b + CH, W + oCK, CBN, CH, CH,
             0, 0, 0, CH, CH, CH, 1.f, 1, 0, 1);
        gemm(W + oMEM, in_w + 2LL * CH * CH, in_b + 2 * CH, W + oCV, CBN, CH, CH,
             0, 0, 0, CH, CH, CH, 1.f, 1, 0, 1);
        cross_score_kernel<<<CB * CHEADS * CNI, 256, 0, stream>>>(W + oCQ, W + oCK, W + oCS);
        cross_attv_kernel<<<blocks((long long)CB * CNI * CH), 256, 0, stream>>>(W + oCS, W + oCV, W + oCO);
        small_linear_kernel<<<blocks((long long)CB * CNI * CH), 256, 0, stream>>>(
            W + oCO, cow, cob, W + oCP, CB * CNI, CH, CH, 0);

        ln_kernel<<<dim3(CB * CNI, 1), 256, 0, stream>>>(W + oDH1, 0, W + oCP, 0, n2g, n2b, W + oDH2, 0);

        // FFN (tiny rows -> VALU)
        small_linear_kernel<<<blocks((long long)CB * CNI * CH), 256, 0, stream>>>(
            W + oDH2, w1, b1, W + oDF1, CB * CNI, CH, CH, 1);
        small_linear_kernel<<<blocks((long long)CB * CNI * CH), 256, 0, stream>>>(
            W + oDF1, w2, b2, W + oDF2, CB * CNI, CH, CH, 0);
        ln_kernel<<<dim3(CB * CNI, 1), 256, 0, stream>>>(W + oDH2, 0, W + oDF2, 0, n3g, n3b, W + oHS, 0);
    }

    // ------------------------- pooling + classifier -------------------------
    pool_cls_kernel<<<CB, 512, 0, stream>>>(W + oHS, P(54), P(55), P(56), P(57), (float*)d_out);
}
